// GNN_Bet4_18485539242351
// MI455X (gfx1250) — compile-verified, matrix operand order checked by reading, not run on hardware
//
#include <hip/hip_runtime.h>
#include <hip/hip_bf16.h>

// ---------------------------------------------------------------------------
// GNN_Bet forward for MI455X (gfx1250, wave32, WMMA).
// - Dense math on v_wmma_f32_16x16x32_bf16 (bf16 in, f32 accumulate),
//   dual accumulators to break WMMA->WMMA RAW hazards (no v_nop fill).
// - Weights pre-packed per-call into WMMA B-fragment-major layout: each
//   fragment is one contiguous 32B vector load per lane (2x global_load_b128).
// - spMM via global_atomic_add_f32 scatter (x = 25.6MB fits in 192MB L2).
// - MLP (128->256->256->1) fully fused per 16-row tile in LDS.
// ---------------------------------------------------------------------------

typedef __bf16  v16bf __attribute__((ext_vector_type(16)));
typedef float   v8f   __attribute__((ext_vector_type(8)));

#define HID   128
#define TILE  16

static __device__ inline __bf16 tobf(float f) { return (__bf16)f; }

// Fragment element e (= 2v+h) -> K offset inside a 32-wide K block.
// e in [0,8):  K = e + 8*g ;  e in [8,16): K = 16 + (e-8) + 8*g
// (ISA 7.12.2 16-bit A/B layout, wave32.)
static __device__ inline v16bf load_afrag(const __bf16* __restrict__ rowp,
                                          int kb, int g) {
  v16bf a;
  const __bf16* p0 = rowp + kb + 8 * g;
  const __bf16* p1 = rowp + kb + 16 + 8 * g;
#pragma unroll
  for (int e = 0; e < 8; ++e) { a[e] = p0[e]; a[8 + e] = p1[e]; }
  return a;
}

// Stage a 16 x HID fp32 tile into LDS as bf16 (float4 loads, 2 per thread).
static __device__ inline void stage_tile(const float* __restrict__ X,
                                         __bf16* __restrict__ xs,
                                         int tileBase, int N, int tid) {
  for (int i = tid; i < TILE * (HID / 4); i += 256) {
    int rr  = i >> 5;           // / (HID/4)
    int c4  = i & 31;
    int row = tileBase + rr;
    float4 p = make_float4(0.f, 0.f, 0.f, 0.f);
    if (row < N) p = ((const float4*)(X + (size_t)row * HID))[c4];
    __bf16* d = xs + rr * HID + c4 * 4;
    d[0] = tobf(p.x); d[1] = tobf(p.y); d[2] = tobf(p.z); d[3] = tobf(p.w);
  }
}

// -------------------------------- utility ----------------------------------

__global__ __launch_bounds__(256) void fill0_kernel(float* __restrict__ p, long n) {
  long i = (long)blockIdx.x * blockDim.x + threadIdx.x;
  if (i < n) p[i] = 0.0f;
}

__global__ __launch_bounds__(256) void mul_kernel(const float* __restrict__ a,
                                                  const float* __restrict__ b,
                                                  float* __restrict__ o, int n) {
  int i = blockIdx.x * blockDim.x + threadIdx.x;
  if (i < n) o[i] = a[i] * b[i];
}

// ----------------------- weight pack: fp32 -> bf16 frags --------------------
// W: K x M row-major fp32.  For frag f = (ct*(K/32)+kb)*32+lane,
// Wp[f][e] = bf16( W[ kb*32 + kmap(e,g) ][ ct*16 + (lane&15) ] ).
__global__ __launch_bounds__(256) void pack_w_kernel(const float* __restrict__ W,
                                                     __bf16* __restrict__ Wp,
                                                     int K, int M) {
  int idx   = blockIdx.x * blockDim.x + threadIdx.x;
  int kblk  = K >> 5;
  int total = (M >> 4) * kblk * 32;
  if (idx >= total) return;
  int lane = idx & 31;
  int kb   = (idx >> 5) % kblk;
  int ct   = (idx >> 5) / kblk;
  int g    = lane >> 4;
  int n    = ct * 16 + (lane & 15);
  v16bf frag;
#pragma unroll
  for (int e = 0; e < 16; ++e) {
    int koff = (e < 8) ? (e + 8 * g) : (16 + (e - 8) + 8 * g);
    frag[e] = tobf(W[(size_t)(kb * 32 + koff) * M + n]);
  }
  *(v16bf*)(Wp + (size_t)idx * 16) = frag;
}

// ------------------------------- spMM scatter ------------------------------
// One wave per edge: 32 lanes x float4 = 128 features.
__global__ __launch_bounds__(256) void spmm_kernel(const int* __restrict__ row,
                                                   const int* __restrict__ col,
                                                   const float* __restrict__ val,
                                                   const float* __restrict__ x,
                                                   float* __restrict__ y, int E) {
  int w    = (int)((blockIdx.x * (long)blockDim.x + threadIdx.x) >> 5);
  int lane = threadIdx.x & 31;
  if (w >= E) return;
  int   r = row[w];
  int   c = col[w];
  float v = val[w];
  float4 p = ((const float4*)(x + (size_t)c * HID))[lane];
  float* yb = y + (size_t)r * HID + lane * 4;
  atomicAdd(yb + 0, v * p.x);
  atomicAdd(yb + 1, v * p.y);
  atomicAdd(yb + 2, v * p.z);
  atomicAdd(yb + 3, v * p.w);
}

// --------------------------- relu (+ row l2norm) ---------------------------
__global__ __launch_bounds__(256) void relunorm_kernel(const float* __restrict__ y,
                                                       float* __restrict__ x,
                                                       int N, int do_norm) {
  int r    = blockIdx.x * 8 + (threadIdx.x >> 5);
  int lane = threadIdx.x & 31;
  if (r >= N) return;
  float4 p = ((const float4*)(y + (size_t)r * HID))[lane];
  p.x = fmaxf(p.x, 0.0f); p.y = fmaxf(p.y, 0.0f);
  p.z = fmaxf(p.z, 0.0f); p.w = fmaxf(p.w, 0.0f);
  if (do_norm) {
    float s = p.x * p.x + p.y * p.y + p.z * p.z + p.w * p.w;
    for (int off = 16; off; off >>= 1) s += __shfl_xor(s, off);
    float inv = 1.0f / fmaxf(sqrtf(s), 1e-12f);
    p.x *= inv; p.y *= inv; p.z *= inv; p.w *= inv;
  }
  ((float4*)(x + (size_t)r * HID))[lane] = p;
}

// ------------------------- layer GEMM: C = X @ W ---------------------------
// X: N x 128 fp32, Wp: packed 128x128 bf16 fragments, C: N x 128 fp32.
// 8 waves / block; wave w owns column tile w; one 16-row tile per block.
__global__ __launch_bounds__(256) void gemm128_kernel(const float* __restrict__ X,
                                                      const __bf16* __restrict__ W,
                                                      float* __restrict__ C, int N) {
  __shared__ __bf16 xs[TILE * HID];
  int tid  = threadIdx.x;
  int wave = tid >> 5, lane = tid & 31;
  int m    = lane & 15, g = lane >> 4;
  int tileBase = (int)blockIdx.x * TILE;
  const v16bf* Wp = (const v16bf*)W;

  // warm L0 with this wave's weight panel (global_prefetch_b8)
  __builtin_prefetch(&Wp[((size_t)wave * 4) * 32 + lane], 0, 3);

  stage_tile(X, xs, tileBase, N, tid);
  __syncthreads();

  const __bf16* arow = xs + m * HID;
  v8f acc0 = {}, acc1 = {};
#pragma unroll
  for (int kbi = 0; kbi < 4; kbi += 2) {
    v16bf a0 = load_afrag(arow, kbi * 32, g);
    v16bf b0 = Wp[((size_t)wave * 4 + kbi) * 32 + lane];
    v16bf a1 = load_afrag(arow, (kbi + 1) * 32, g);
    v16bf b1 = Wp[((size_t)wave * 4 + kbi + 1) * 32 + lane];
    acc0 = __builtin_amdgcn_wmma_f32_16x16x32_bf16(false, a0, false, b0,
                                                   (short)0, acc0, false, false);
    acc1 = __builtin_amdgcn_wmma_f32_16x16x32_bf16(false, a1, false, b1,
                                                   (short)0, acc1, false, false);
  }

  int n = wave * 16 + m;
  float* cp = C + (size_t)tileBase * HID + n;
  if (tileBase + TILE <= N) {
#pragma unroll
    for (int r = 0; r < 8; ++r) cp[(r + 8 * g) * HID] = acc0[r] + acc1[r];
  } else {
#pragma unroll
    for (int r = 0; r < 8; ++r)
      if (tileBase + r + 8 * g < N) cp[(r + 8 * g) * HID] = acc0[r] + acc1[r];
  }
}

// ----------------------------- fused MLP + dot -----------------------------
// score[i] += relu(relu(x@W1+b1)@W2+b2) @ W3 + b3, one block per 16-row tile.
// W1p: packed 128x256, W2p: packed 256x256 (fragment-major bf16).
__global__ __launch_bounds__(256) void mlp_fused_kernel(const float*  __restrict__ X,
                                                        const __bf16* __restrict__ W1,
                                                        const float*  __restrict__ B1,
                                                        const __bf16* __restrict__ W2,
                                                        const float*  __restrict__ B2,
                                                        const float*  __restrict__ W3,
                                                        const float*  __restrict__ B3,
                                                        float* __restrict__ score, int N) {
  __shared__ __bf16 xs [TILE * HID];   // 16 x 128 bf16
  __shared__ __bf16 h1s[TILE * 256];   // 16 x 256 bf16
  __shared__ float  sdot[TILE];

  int tid  = threadIdx.x;
  int wave = tid >> 5, lane = tid & 31;
  int m    = lane & 15, g = lane >> 4;
  int tileBase = (int)blockIdx.x * TILE;
  const v16bf* W1p = (const v16bf*)W1;
  const v16bf* W2p = (const v16bf*)W2;

  __builtin_prefetch(&W1p[((size_t)wave * 4) * 32 + lane], 0, 3);

  stage_tile(X, xs, tileBase, N, tid);
  if (tid < TILE) sdot[tid] = 0.0f;
  __syncthreads();

  // ---- layer 1: h1 = relu(x @ W1 + b1), W1: 128x256 ----
  const __bf16* arow = xs + m * HID;
#pragma unroll
  for (int cc = 0; cc < 2; ++cc) {
    int ct = wave + 8 * cc;
    int n  = ct * 16 + m;
    v8f acc0 = {}, acc1 = {};
#pragma unroll
    for (int kbi = 0; kbi < 4; kbi += 2) {
      v16bf a0 = load_afrag(arow, kbi * 32, g);
      v16bf b0 = W1p[((size_t)ct * 4 + kbi) * 32 + lane];
      v16bf a1 = load_afrag(arow, (kbi + 1) * 32, g);
      v16bf b1 = W1p[((size_t)ct * 4 + kbi + 1) * 32 + lane];
      acc0 = __builtin_amdgcn_wmma_f32_16x16x32_bf16(false, a0, false, b0,
                                                     (short)0, acc0, false, false);
      acc1 = __builtin_amdgcn_wmma_f32_16x16x32_bf16(false, a1, false, b1,
                                                     (short)0, acc1, false, false);
    }
    float bias = B1[n];
#pragma unroll
    for (int r = 0; r < 8; ++r)
      h1s[(r + 8 * g) * 256 + n] = tobf(fmaxf(acc0[r] + acc1[r] + bias, 0.0f));
  }
  __syncthreads();

  // ---- layer 2 + layer 3 fused: sdot += relu(h1@W2+b2) . W3 ----
  const __bf16* hrow = h1s + m * 256;
#pragma unroll
  for (int cc = 0; cc < 2; ++cc) {
    int ct = wave + 8 * cc;
    int n  = ct * 16 + m;
    v8f acc0 = {}, acc1 = {};
#pragma unroll
    for (int kbi = 0; kbi < 8; kbi += 2) {
      v16bf a0 = load_afrag(hrow, kbi * 32, g);
      v16bf b0 = W2p[((size_t)ct * 8 + kbi) * 32 + lane];
      v16bf a1 = load_afrag(hrow, (kbi + 1) * 32, g);
      v16bf b1 = W2p[((size_t)ct * 8 + kbi + 1) * 32 + lane];
      acc0 = __builtin_amdgcn_wmma_f32_16x16x32_bf16(false, a0, false, b0,
                                                     (short)0, acc0, false, false);
      acc1 = __builtin_amdgcn_wmma_f32_16x16x32_bf16(false, a1, false, b1,
                                                     (short)0, acc1, false, false);
    }
    float bias = B2[n];
    float w3   = W3[n];
#pragma unroll
    for (int r = 0; r < 8; ++r) {
      float p = fmaxf(acc0[r] + acc1[r] + bias, 0.0f) * w3;
      p += __shfl_xor(p, 1);
      p += __shfl_xor(p, 2);
      p += __shfl_xor(p, 4);
      p += __shfl_xor(p, 8);
      if (m == 0) atomicAdd(&sdot[r + 8 * g], p);  // ds_add_f32
    }
  }
  __syncthreads();

  if (tid < TILE) {
    int row = tileBase + tid;
    if (row < N) score[row] += sdot[tid] + B3[0];
  }
}

// ------------------------------ orchestration ------------------------------

extern "C" void kernel_launch(void* const* d_in, const int* in_sizes, int n_in,
                              void* d_out, int out_size, void* d_ws, size_t ws_size,
                              hipStream_t stream) {
  const int E = in_sizes[0];
  const int N = in_sizes[6] / HID;

  const int*   rows[2] = {(const int*)d_in[0], (const int*)d_in[3]};
  const int*   cols[2] = {(const int*)d_in[1], (const int*)d_in[4]};
  const float* vals[2] = {(const float*)d_in[2], (const float*)d_in[5]};
  const float* W0     = (const float*)d_in[6];
  const float* WgF[4] = {(const float*)d_in[7], (const float*)d_in[8],
                         (const float*)d_in[9], (const float*)d_in[10]};
  const float* L1w = (const float*)d_in[11];
  const float* L1b = (const float*)d_in[12];
  const float* L2w = (const float*)d_in[13];
  const float* L2b = (const float*)d_in[14];
  const float* L3w = (const float*)d_in[15];
  const float* L3b = (const float*)d_in[16];

  // workspace layout (~77.5 MB), all chunks 32B-aligned
  char* w = (char*)d_ws;
  float* x  = (float*)w; w += (size_t)N * HID * 4;
  float* y  = (float*)w; w += (size_t)N * HID * 4;
  float* t  = (float*)w; w += (size_t)N * HID * 4;
  float* s1 = (float*)w; w += (size_t)N * 4;
  float* s2 = (float*)w; w += (size_t)N * 4;
  __bf16* wgb[4];
  for (int i = 0; i < 4; ++i) { wgb[i] = (__bf16*)w; w += (size_t)HID * HID * 2; }
  __bf16* l1wb = (__bf16*)w;  w += (size_t)HID * 256 * 2;
  __bf16* l2wb = (__bf16*)w;  w += (size_t)256 * 256 * 2;
  (void)ws_size; (void)n_in; (void)out_size;

  const int tiles = (N + TILE - 1) / TILE;

  // 1) weights -> packed bf16 fragments (deterministic, every call)
  for (int i = 0; i < 4; ++i)
    pack_w_kernel<<<(1024 + 255) / 256, 256, 0, stream>>>(WgF[i], wgb[i], 128, 128);
  pack_w_kernel<<<(2048 + 255) / 256, 256, 0, stream>>>(L1w, l1wb, 128, 256);
  pack_w_kernel<<<(4096 + 255) / 256, 256, 0, stream>>>(L2w, l2wb, 256, 256);

  // 2) zero both score accumulators (contiguous s1|s2)
  fill0_kernel<<<(int)(((long)2 * N + 255) / 256), 256, 0, stream>>>(s1, (long)2 * N);

  const long nh    = (long)N * HID;
  const int  fillB = (int)((nh + 255) / 256);
  const int  spmmB = (E + 7) / 8;   // 8 waves per 256-thread block
  const int  normB = (N + 7) / 8;

  float* sc[2] = {s1, s2};
  for (int b = 0; b < 2; ++b) {
    for (int layer = 0; layer < 5; ++layer) {
      const float* spin;
      if (layer == 0) {
        spin = W0;
      } else {
        gemm128_kernel<<<tiles, 256, 0, stream>>>(x, wgb[layer - 1], t, N);
        spin = t;
      }
      fill0_kernel<<<fillB, 256, 0, stream>>>(y, nh);
      spmm_kernel<<<spmmB, 256, 0, stream>>>(rows[b], cols[b], vals[b], spin, y, E);
      relunorm_kernel<<<normB, 256, 0, stream>>>(y, x, N, layer < 4 ? 1 : 0);
      mlp_fused_kernel<<<tiles, 256, 0, stream>>>(x, l1wb, L1b, l2wb, L2b,
                                                  L3w, L3b, sc[b], N);
    }
  }

  // 3) out = s1 * s2
  mul_kernel<<<(N + 255) / 256, 256, 0, stream>>>(s1, s2, (float*)d_out, N);
}
